// ShieldingT2Model_88175678587466
// MI455X (gfx1250) — matrix operand order repaired and checked by reading, not compiled, not run on hardware
//
#include <hip/hip_runtime.h>
#include <hip/hip_bf16.h>
#include <math.h>

typedef __attribute__((ext_vector_type(16))) __bf16 v16bf;
typedef __attribute__((ext_vector_type(8)))  float  v8f;

#define DEV __device__ __forceinline__

DEV __bf16 f2bf(float f) { return (__bf16)f; }   // native v_cvt (RTNE)
DEV float  bf2f(__bf16 b) { return (float)b; }

DEV v8f wmma_bf(v16bf a, v16bf b, v8f c) {
  return __builtin_amdgcn_wmma_f32_16x16x32_bf16(false, a, false, b, (short)0, c, false, false);
}

DEV float silu_act(float z) {          // 1.679 * z * sigmoid(z)
  float e = __builtin_amdgcn_exp2f(z * -1.44269504f);
  return 1.679f * z * __builtin_amdgcn_rcpf(1.f + e);
}

// --- wave32 WMMA fragment index helpers (CDNA5 ISA 7.12.2) ---
DEV int a_kidx(int lane, int j) { return ((lane >> 4) * 16) + j; }
DEV int b_kofs(int lane) { return (lane >> 4) * 16; }

struct U32x8 { uint4 a, b; };

// A fragment from bf16 LDS buffer: two contiguous 16B chunks per lane
DEV v16bf a_from_bf(const __bf16* A, int ld, int kbase, int lane) {
  const __bf16* p = A + (lane & 15) * ld + kbase + ((lane >> 4) * 16);
  U32x8 t;
  t.a = *(const uint4*)p;
  t.b = *(const uint4*)(p + 8);
  return __builtin_bit_cast(v16bf, t);
}
// B fragment from pre-packed bf16 global buffer: layout [ktile][col][k%32]
// -> per lane one contiguous 32B run = two global_load_b128
DEV v16bf b_packed(const __bf16* PB, int N, int kt, int col, int lane) {
  const __bf16* p = PB + ((size_t)kt * N + col) * 32 + ((lane >> 4) * 16);
  U32x8 t;
  t.a = *(const uint4*)p;
  t.b = *(const uint4*)(p + 8);
  return __builtin_bit_cast(v16bf, t);
}
// Fallback gather loaders (f32 weights straight from global)
DEV v16bf b_row(const float* B, int ld, int kbase, int col, int lane) {
  v16bf b; int kb = kbase + b_kofs(lane);
#pragma unroll
  for (int j = 0; j < 16; ++j) b[j] = f2bf(B[(size_t)(kb + j) * ld + col]);
  return b;
}
DEV v16bf b_colT(const float* B, int ld, int kbase, int col, int lane) {
  v16bf b; int kb = kbase + b_kofs(lane);
#pragma unroll
  for (int j = 0; j < 16; ++j) b[j] = f2bf(B[(size_t)col * ld + kb + j]);
  return b;
}

// --- D stores (D: M = p + 8*(lane>=16), N = lane&15) ---
DEV void store_d_f32(float* D, int ld, int ntile, v8f acc, int lane, float scale) {
  int n = ntile * 16 + (lane & 15); int mb = (lane >> 4) * 8;
#pragma unroll
  for (int p = 0; p < 8; ++p) D[(mb + p) * ld + n] = acc[p] * scale;
}
DEV void store_d_bf(__bf16* D, int ld, int ntile, v8f acc, int lane, float scale) {
  int n = ntile * 16 + (lane & 15); int mb = (lane >> 4) * 8;
#pragma unroll
  for (int p = 0; p < 8; ++p) D[(mb + p) * ld + n] = f2bf(acc[p] * scale);
}
DEV void store_act(__bf16* D, int ld, int ntile, v8f acc, int lane, float scale) {
  int n = ntile * 16 + (lane & 15); int mb = (lane >> 4) * 8;
#pragma unroll
  for (int p = 0; p < 8; ++p) D[(mb + p) * ld + n] = f2bf(silu_act(acc[p] * scale));
}

// --- Wigner C222 (real basis, unit Frobenius norm, sign-fixed like reference) ---
DEV float C222f(int i, int j, int k) {
  int a = i, b = j, c = k, t;
  if (a > b) { t = a; a = b; b = t; }
  if (b > c) { t = b; b = c; c = t; }
  if (a > b) { t = a; a = b; b = t; }
  const float C1 = 0.23904572f, C2 = 0.11952286f, C3 = 0.20701967f;
  if (a == 0 && b == 0 && c == 2) return  C1;
  if (a == 2 && b == 4 && c == 4) return  C1;
  if (a == 2 && b == 2 && c == 2) return -C1;
  if (a == 1 && b == 1 && c == 2) return -C2;
  if (a == 2 && b == 3 && c == 3) return -C2;
  if (a == 1 && b == 1 && c == 4) return  C3;
  if (a == 3 && b == 3 && c == 4) return -C3;
  if (a == 0 && b == 1 && c == 3) return -C3;
  return 0.f;
}

// A features generated on the fly ---------------------------------------
DEV v16bf a_outer(const float* S, int kbase, int lane) {   // s_u * s_v
  const float* sr = S + (lane & 15) * 64;
  v16bf a;
#pragma unroll
  for (int j = 0; j < 16; ++j) {
    int kk = kbase + a_kidx(lane, j);
    a[j] = f2bf(sr[kk >> 6] * sr[kk & 63]);
  }
  return a;
}
DEV v16bf a_q(const float* T, int kbase, int lane) {  // (1/sqrt5) sum_i t_ui t_vi
  const float* tr = T + (lane & 15) * 80;
  v16bf a;
#pragma unroll
  for (int j = 0; j < 16; ++j) {
    int kk = kbase + a_kidx(lane, j); int u = kk >> 4, v = kk & 15;
    float acc = 0.f;
#pragma unroll
    for (int i = 0; i < 5; ++i) acc += tr[u * 5 + i] * tr[v * 5 + i];
    a[j] = f2bf(acc * 0.44721360f);
  }
  return a;
}

// ---- packed-weight offsets in d_ws (bf16 elements) ----
#define O_W1    0u
#define O_W2    16384u
#define O_W3    81920u
#define O_W000  90112u
#define O_W220  614400u
#define O_W022  647168u
#define O_W202  778240u
#define O_W222  909312u
#define O_W2A   942080u
#define O_W2B   958464u
#define O_W2C   974848u
#define PACK_TOTAL 991232u     // elements; *2 bytes = ~1.94 MB

// Repack f32 weight (K x N view) into bf16 B-fragment layout [ktile][col][k%32].
// mode 0: src[k*N + c]   mode 1 (transposed): src[c*K + k]
// mode 2 (w202 view):    src[(c>>7)*8192 + k*128 + (c&127)]
__global__ void repack_b_kernel(const float* __restrict__ src, __bf16* __restrict__ dst,
                                int K, int N, int mode) {
  int total = K * N;
  for (int i = blockIdx.x * blockDim.x + threadIdx.x; i < total;
       i += gridDim.x * blockDim.x) {
    int k = i / N, c = i % N;
    float v;
    if (mode == 0)      v = src[(size_t)k * N + c];
    else if (mode == 1) v = src[(size_t)c * K + k];
    else                v = src[(size_t)(c >> 7) * 8192 + (size_t)k * 128 + (c & 127)];
    dst[((size_t)(k >> 5) * N + c) * 32 + (k & 31)] = f2bf(v);
  }
}

// ---- LDS pool layout (bytes) ----
#define OFF_S     0        // f32  16x64
#define OFF_SB    4096     // bf16 16x64
#define OFF_T     6144     // f32  16x80
#define OFF_REL   11264    // f32  16x32
#define OFF_MIX   13312    // f32  16x5 (320B)
#define OFF_Y0    13632    // bf16 16x128
#define OFF_G     17728    // f32  16x128
#define OFF_TMP   25920    // bf16 16x256 (E-features / chunk outs; alias f32 16x128 MF)
#define OFF_Y2    34112    // bf16 16x640 (H1/H2 alias this region during MLP)
#define POOL_SZ   54656

template<bool PACKED>
__global__ __launch_bounds__(32, 1)
void shielding_t2_fused(
    const float* __restrict__ scal, const float* __restrict__ t2s,
    const float* __restrict__ w1, const float* __restrict__ w2, const float* __restrict__ w3,
    const float* __restrict__ thr,
    const float* __restrict__ w000, const float* __restrict__ w220,
    const float* __restrict__ w022, const float* __restrict__ w202, const float* __restrict__ w222,
    const float* __restrict__ w2a, const float* __restrict__ w2b, const float* __restrict__ w2c,
    const float* __restrict__ csp, const __bf16* __restrict__ pk, float* __restrict__ out)
{
  __shared__ __align__(16) char pool[POOL_SZ];
  float*   S   = (float*)(pool + OFF_S);
  __bf16*  SB  = (__bf16*)(pool + OFF_SB);
  float*   T   = (float*)(pool + OFF_T);
  float*   REL = (float*)(pool + OFF_REL);
  float*   MIX = (float*)(pool + OFF_MIX);
  __bf16*  Y0  = (__bf16*)(pool + OFF_Y0);
  float*   G   = (float*)(pool + OFF_G);
  __bf16*  TMP = (__bf16*)(pool + OFF_TMP);
  float*   MF  = (float*)(pool + OFF_TMP);
  __bf16*  Y2  = (__bf16*)(pool + OFF_Y2);
  __bf16*  H1  = (__bf16*)(pool + OFF_Y2);
  __bf16*  H2  = (__bf16*)(pool + OFF_Y2 + 8192);

  const float A0  = 0.01515847f;   // sqrt(1/(64^2+16^2))
  const float A2  = 0.04658475f;   // sqrt(5/(2*64*16+16^2))
  const float B2  = 0.01008589f;   // sqrt(5/(3*128*128))
  const float IS5 = 0.44721360f;   // 1/sqrt(5)

  const int lane = threadIdx.x;
  const int row0 = blockIdx.x * 16;

  // ---- stage scalars (+bf16 copy) and build t = [t2s[:,0:15,:], sum_k t2s] ----
  for (int idx = lane; idx < 16 * 64; idx += 32) {
    float v = scal[(size_t)(row0 + (idx >> 6)) * 64 + (idx & 63)];
    S[idx] = v; SB[idx] = f2bf(v);
  }
  for (int idx = lane; idx < 16 * 80; idx += 32) {
    int r = idx / 80, c = idx % 80, u = c / 5, i = c % 5;
    const float* base = t2s + (size_t)(row0 + r) * 160;
    float v;
    if (u < 15) v = base[u * 5 + i];
    else { v = 0.f; for (int k = 0; k < 32; ++k) v += base[k * 5 + i]; }
    T[idx] = v;
  }
  __syncthreads();

  // generic B-fragment fetch: packed bf16 path or f32 gather fallback
  auto BF = [&](const float* W, unsigned po, int N, int kt, int col) -> v16bf {
    if constexpr (PACKED) return b_packed(pk + po, N, kt, col, lane);
    else return b_row(W, N, kt * 32, col, lane);
  };

  // ---- MLP: h1 = act(S@W1/8) ----
#pragma unroll 1
  for (int n = 0; n < 16; ++n) {
    v8f acc = {0.f,0.f,0.f,0.f,0.f,0.f,0.f,0.f};
#pragma unroll
    for (int kt = 0; kt < 2; ++kt)
      acc = wmma_bf(a_from_bf(SB, 64, kt * 32, lane),
                    BF(w1, O_W1, 256, kt, n * 16 + (lane & 15)), acc);
    store_act(H1, 256, n, acc, lane, 0.125f);
  }
  __syncthreads();
  // h2 = act(h1@W2/16)
#pragma unroll 1
  for (int n = 0; n < 16; ++n) {
    v8f acc = {0.f,0.f,0.f,0.f,0.f,0.f,0.f,0.f};
#pragma unroll 1
    for (int kt = 0; kt < 8; ++kt)
      acc = wmma_bf(a_from_bf(H1, 256, kt * 32, lane),
                    BF(w2, O_W2, 256, kt, n * 16 + (lane & 15)), acc);
    store_act(H2, 256, n, acc, lane, 0.0625f);
  }
  __syncthreads();
  // rel = h2@W3/16
#pragma unroll 1
  for (int n = 0; n < 2; ++n) {
    v8f acc = {0.f,0.f,0.f,0.f,0.f,0.f,0.f,0.f};
#pragma unroll 1
    for (int kt = 0; kt < 8; ++kt)
      acc = wmma_bf(a_from_bf(H2, 256, kt * 32, lane),
                    BF(w3, O_W3, 32, kt, n * 16 + (lane & 15)), acc);
    store_d_f32(REL, 32, n, acc, lane, 0.0625f);
  }
  __syncthreads();

  // ---- gate + mixed (lanes 0-15 = rows) ----
  if (lane < 16) {
    const float* base = t2s + (size_t)(row0 + lane) * 160;
    float mix[5] = {0.f,0.f,0.f,0.f,0.f};
    for (int k = 0; k < 32; ++k) {
      float e[5], m2 = 0.f;
#pragma unroll
      for (int i = 0; i < 5; ++i) { e[i] = base[k * 5 + i]; m2 += e[i] * e[i]; }
      float mag = sqrtf(m2);
      float g = mag / (mag + thr[k]);
      float w = REL[lane * 32 + k] * g * 0.17677670f;   // 1/sqrt(32)
#pragma unroll
      for (int i = 0; i < 5; ++i) mix[i] += w * e[i];
    }
#pragma unroll
    for (int i = 0; i < 5; ++i) MIX[lane * 5 + i] = mix[i];
  }
  __syncthreads();

  // ---- y0 = a0*( outer(s,s)@W000 + q(t)@W220 ); K-outer, 8 N accumulators ----
  {
    v8f acc[8];
#pragma unroll
    for (int n = 0; n < 8; ++n) acc[n] = (v8f){0.f,0.f,0.f,0.f,0.f,0.f,0.f,0.f};
#pragma unroll 1
    for (int kt = 0; kt < 128; ++kt) {
      if constexpr (PACKED)
        __builtin_prefetch((const void*)(pk + O_W000 + (size_t)(((kt + 8) & 127) * 32) * 128), 0, 1);
      else
        __builtin_prefetch((const void*)(w000 + (size_t)(((kt + 8) & 127) * 32) * 128), 0, 1);
      v16bf a = a_outer(S, kt * 32, lane);
#pragma unroll
      for (int n = 0; n < 8; ++n)
        acc[n] = wmma_bf(a, BF(w000, O_W000, 128, kt, n * 16 + (lane & 15)), acc[n]);
    }
#pragma unroll 1
    for (int kt = 0; kt < 8; ++kt) {
      v16bf a = a_q(T, kt * 32, lane);
#pragma unroll
      for (int n = 0; n < 8; ++n)
        acc[n] = wmma_bf(a, BF(w220, O_W220, 128, kt, n * 16 + (lane & 15)), acc[n]);
    }
#pragma unroll
    for (int n = 0; n < 8; ++n) store_d_bf(Y0, 128, n, acc[n], lane, A0);
  }
  __syncthreads();

  // ---- y2 pass 1 (w222): per k build E-features once in LDS, GEMM all 8 h-tiles ----
#pragma unroll 1
  for (int k = 0; k < 5; ++k) {
    float Ck[25];
#pragma unroll
    for (int ii = 0; ii < 5; ++ii)
#pragma unroll
      for (int jj = 0; jj < 5; ++jj) Ck[ii * 5 + jj] = C222f(ii, jj, k);
    for (int idx = lane; idx < 16 * 256; idx += 32) {
      int r = idx >> 8, kk = idx & 255, u = kk >> 4, v = kk & 15;
      const float* tr = T + r * 80;
      float acc = 0.f;
#pragma unroll
      for (int ii = 0; ii < 5; ++ii)
#pragma unroll
        for (int jj = 0; jj < 5; ++jj)
          acc += Ck[ii * 5 + jj] * tr[u * 5 + ii] * tr[v * 5 + jj];
      TMP[idx] = f2bf(acc);
    }
    __syncthreads();
    v8f acc[8];
#pragma unroll
    for (int n = 0; n < 8; ++n) acc[n] = (v8f){0.f,0.f,0.f,0.f,0.f,0.f,0.f,0.f};
#pragma unroll 1
    for (int kt = 0; kt < 8; ++kt) {
      v16bf a = a_from_bf(TMP, 256, kt * 32, lane);
#pragma unroll
      for (int nh = 0; nh < 8; ++nh)
        acc[nh] = wmma_bf(a, BF(w222, O_W222, 128, kt, nh * 16 + (lane & 15)), acc[nh]);
    }
    int hh = lane & 15, mb = (lane >> 4) * 8;
#pragma unroll
    for (int nh = 0; nh < 8; ++nh)
#pragma unroll
      for (int p = 0; p < 8; ++p)
        Y2[(mb + p) * 640 + k * 128 + nh * 16 + hh] = f2bf(A2 * acc[nh][p]);
    __syncthreads();
  }

  // ---- y2 pass 2: 022/202 factored paths, accumulated into Y2 (rmw) ----
#pragma unroll 1
  for (int nh = 0; nh < 8; ++nh) {
    // A022[b,v,h] = sum_u W022[u,v,h] s_u  (chunk cols (v,hh))
#pragma unroll 1
    for (int n2 = 0; n2 < 16; ++n2) {
      v8f acc = {0.f,0.f,0.f,0.f,0.f,0.f,0.f,0.f};
      int col = n2 * 16 + (lane & 15), v = col >> 4, hh = col & 15;
#pragma unroll
      for (int kt = 0; kt < 2; ++kt) {
        v16bf b;
        if constexpr (PACKED) {
          b = b_packed(pk + O_W022, 2048, kt, v * 128 + nh * 16 + hh, lane);
        } else {
          int kb = kt * 32 + b_kofs(lane);
#pragma unroll
          for (int j = 0; j < 16; ++j)
            b[j] = f2bf(w022[(size_t)(kb + j) * 2048 + v * 128 + nh * 16 + hh]);
        }
        acc = wmma_bf(a_from_bf(SB, 64, kt * 32, lane), b, acc);
      }
      store_d_bf(TMP, 256, n2, acc, lane, 1.0f);
    }
    __syncthreads();
    { // Y2[r][k][nh*16+hh] += a2/sqrt5 * sum_v TMP[r][v*16+hh] * t[r][v*5+k]
      int hh = lane & 15, rb = (lane >> 4) * 8;
      for (int r = rb; r < rb + 8; ++r) {
        float av[5] = {0.f,0.f,0.f,0.f,0.f};
        for (int v = 0; v < 16; ++v) {
          float mv = bf2f(TMP[r * 256 + v * 16 + hh]);
#pragma unroll
          for (int k = 0; k < 5; ++k) av[k] += mv * T[r * 80 + v * 5 + k];
        }
#pragma unroll
        for (int k = 0; k < 5; ++k) {
          int o = r * 640 + k * 128 + nh * 16 + hh;
          Y2[o] = f2bf(bf2f(Y2[o]) + av[k] * (A2 * IS5));
        }
      }
    }
    __syncthreads();
    // D202[b,u,h] = sum_v W202[u,v,h] s_v  (chunk cols (u,hh))
#pragma unroll 1
    for (int n2 = 0; n2 < 16; ++n2) {
      v8f acc = {0.f,0.f,0.f,0.f,0.f,0.f,0.f,0.f};
      int col = n2 * 16 + (lane & 15), u = col >> 4, hh = col & 15;
#pragma unroll
      for (int kt = 0; kt < 2; ++kt) {
        v16bf b;
        if constexpr (PACKED) {
          b = b_packed(pk + O_W202, 2048, kt, u * 128 + nh * 16 + hh, lane);
        } else {
          int kb = kt * 32 + b_kofs(lane);
#pragma unroll
          for (int j = 0; j < 16; ++j)
            b[j] = f2bf(w202[(size_t)u * 8192 + (size_t)(kb + j) * 128 + nh * 16 + hh]);
        }
        acc = wmma_bf(a_from_bf(SB, 64, kt * 32, lane), b, acc);
      }
      store_d_bf(TMP, 256, n2, acc, lane, 1.0f);
    }
    __syncthreads();
    {
      int hh = lane & 15, rb = (lane >> 4) * 8;
      for (int r = rb; r < rb + 8; ++r) {
        float av[5] = {0.f,0.f,0.f,0.f,0.f};
        for (int u = 0; u < 16; ++u) {
          float mv = bf2f(TMP[r * 256 + u * 16 + hh]);
#pragma unroll
          for (int k = 0; k < 5; ++k) av[k] += mv * T[r * 80 + u * 5 + k];
        }
#pragma unroll
        for (int k = 0; k < 5; ++k) {
          int o = r * 640 + k * 128 + nh * 16 + hh;
          Y2[o] = f2bf(bf2f(Y2[o]) + av[k] * (A2 * IS5));
        }
      }
    }
    __syncthreads();
  }

  // ---- tp2: G = y0@Wa + y0@Wb^T  (g+e combined since both dot y2) ----
  {
    v8f acc[8];
#pragma unroll
    for (int n = 0; n < 8; ++n) acc[n] = (v8f){0.f,0.f,0.f,0.f,0.f,0.f,0.f,0.f};
#pragma unroll 1
    for (int kt = 0; kt < 4; ++kt) {
      v16bf a = a_from_bf(Y0, 128, kt * 32, lane);
#pragma unroll
      for (int n = 0; n < 8; ++n) {
        int col = n * 16 + (lane & 15);
        acc[n] = wmma_bf(a, BF(w2a, O_W2A, 128, kt, col), acc[n]);
        v16bf bb;
        if constexpr (PACKED) bb = b_packed(pk + O_W2B, 128, kt, col, lane);
        else                  bb = b_colT(w2b, 128, kt * 32, col, lane);
        acc[n] = wmma_bf(a, bb, acc[n]);
      }
    }
#pragma unroll
    for (int n = 0; n < 8; ++n) store_d_f32(G, 128, n, acc[n], lane, 1.0f);
  }
  __syncthreads();

  float corr[5] = {0.f,0.f,0.f,0.f,0.f};
  if (lane < 16) {
#pragma unroll 1
    for (int k = 0; k < 5; ++k) {
      float a = 0.f;
      for (int h = 0; h < 128; ++h) a += G[lane * 128 + h] * bf2f(Y2[lane * 640 + k * 128 + h]);
      corr[k] = a * IS5;
    }
  }

  // ---- tp2 w222: per i, m_i = y2[:,:,i]@Wc ; corr_c[k] += C[ijk]*(m_i . y2[:,:,j]) ----
#pragma unroll 1
  for (int i = 0; i < 5; ++i) {
    __syncthreads();
    {
      v8f acc[8];
#pragma unroll
      for (int n = 0; n < 8; ++n) acc[n] = (v8f){0.f,0.f,0.f,0.f,0.f,0.f,0.f,0.f};
#pragma unroll 1
      for (int kt = 0; kt < 4; ++kt) {
        v16bf a = a_from_bf(Y2 + i * 128, 640, kt * 32, lane);
#pragma unroll
        for (int n = 0; n < 8; ++n)
          acc[n] = wmma_bf(a, BF(w2c, O_W2C, 128, kt, n * 16 + (lane & 15)), acc[n]);
      }
#pragma unroll
      for (int n = 0; n < 8; ++n) store_d_f32(MF, 128, n, acc[n], lane, 1.0f);
    }
    __syncthreads();
    if (lane < 16) {
#pragma unroll 1
      for (int j = 0; j < 5; ++j) {
        float d = 0.f;
        for (int h = 0; h < 128; ++h) d += MF[lane * 128 + h] * bf2f(Y2[lane * 640 + j * 128 + h]);
#pragma unroll
        for (int k = 0; k < 5; ++k) corr[k] += C222f(i, j, k) * d;
      }
    }
  }
  __syncthreads();

  // ---- output ----
  if (lane < 16) {
    float cs = csp[0];
#pragma unroll
    for (int k = 0; k < 5; ++k)
      out[(size_t)(row0 + lane) * 5 + k] = MIX[lane * 5 + k] + cs * B2 * corr[k];
  }
}

extern "C" void kernel_launch(void* const* d_in, const int* in_sizes, int n_in,
                              void* d_out, int out_size, void* d_ws, size_t ws_size,
                              hipStream_t stream) {
  const float* scal = (const float*)d_in[0];
  const float* t2s  = (const float*)d_in[1];
  const float* w1   = (const float*)d_in[2];
  const float* w2   = (const float*)d_in[3];
  const float* w3   = (const float*)d_in[4];
  const float* thr  = (const float*)d_in[5];
  const float* w000 = (const float*)d_in[6];
  const float* w220 = (const float*)d_in[7];
  const float* w022 = (const float*)d_in[8];
  const float* w202 = (const float*)d_in[9];
  const float* w222 = (const float*)d_in[10];
  const float* w2a  = (const float*)d_in[11];
  const float* w2b  = (const float*)d_in[12];
  const float* w2c  = (const float*)d_in[13];
  const float* csp  = (const float*)d_in[14];
  float* outp = (float*)d_out;

  int batch = in_sizes[0] / 64;          // 16384
  dim3 grid(batch / 16), block(32);

  bool packed = (d_ws != nullptr) && (ws_size >= (size_t)PACK_TOTAL * 2);
  if (packed) {
    __bf16* pw = (__bf16*)d_ws;
    auto rp = [&](const float* src, unsigned ofs, int K, int N, int mode) {
      int total = K * N;
      int blocks = (total + 255) / 256; if (blocks > 512) blocks = 512;
      repack_b_kernel<<<blocks, 256, 0, stream>>>(src, pw + ofs, K, N, mode);
    };
    rp(w1,   O_W1,   64,   256,  0);
    rp(w2,   O_W2,   256,  256,  0);
    rp(w3,   O_W3,   256,  32,   0);
    rp(w000, O_W000, 4096, 128,  0);
    rp(w220, O_W220, 256,  128,  0);
    rp(w022, O_W022, 64,   2048, 0);
    rp(w202, O_W202, 64,   2048, 2);   // src[(c>>7)*8192 + k*128 + (c&127)]
    rp(w222, O_W222, 256,  128,  0);
    rp(w2a,  O_W2A,  128,  128,  0);
    rp(w2b,  O_W2B,  128,  128,  1);   // transposed
    rp(w2c,  O_W2C,  128,  128,  0);
    shielding_t2_fused<true><<<grid, block, 0, stream>>>(
        scal, t2s, w1, w2, w3, thr, w000, w220, w022, w202, w222,
        w2a, w2b, w2c, csp, pw, outp);
  } else {
    shielding_t2_fused<false><<<grid, block, 0, stream>>>(
        scal, t2s, w1, w2, w3, thr, w000, w220, w022, w202, w222,
        w2a, w2b, w2c, csp, nullptr, outp);
  }
}
